// Megnet_EdgeModel_62689342653097
// MI455X (gfx1250) — compile-verified, matrix operand order checked by reading, not run on hardware
//
#include <hip/hip_runtime.h>
#include <hip/hip_bf16.h>
#include <stdint.h>

typedef __attribute__((ext_vector_type(16))) _Float16 v16h;
typedef __attribute__((ext_vector_type(8)))  _Float16 v8h;
typedef __attribute__((ext_vector_type(8)))  float    v8f;
typedef __attribute__((ext_vector_type(4)))  float    v4f;
typedef __attribute__((ext_vector_type(4)))  int      v4i;

#define E_EDGES 500000
#define D_DIM   128
#define EPS_BN  1e-5f

// Padded strides (halves): +16B per row vs the 256B bank period -> a
// 16-lane x 16B fragment load covers all 64 banks exactly once.
#define SW0   520   // W0^T row stride (1040 B)
#define SW12  136   // W1^T/W2^T row stride (272 B)
#define SACT  136   // activation tile row stride (272 B)

// Pre-converted weight image in d_ws (halves):
//   [0, 66560)        W0^T [128][SW0]
//   [66560, 83968)    W1^T [128][SW12]
//   [83968, 101376)   W2^T [128][SW12]
#define WSH_W1    66560
#define WSH_W2    83968
#define WSH_TOTAL 101376                 // halves
#define WS_BYTES  (WSH_TOTAL * 2)        // 202752 B
#define WS_CHUNKS (WS_BYTES / 16)        // 12672 x 16B

static __device__ __forceinline__ v16h cat8(v8h lo, v8h hi) {
    return __builtin_shufflevector(lo, hi, 0,1,2,3,4,5,6,7,8,9,10,11,12,13,14,15);
}

static __device__ __forceinline__ v8f zero8() {
    v8f z;
#pragma unroll
    for (int i = 0; i < 8; ++i) z[i] = 0.0f;
    return z;
}

static __device__ __forceinline__ v16h loadA16(const float* p) {
    const v4f f0 = *(const v4f*)(p);
    const v4f f1 = *(const v4f*)(p + 4);
    const v4f f2 = *(const v4f*)(p + 16);
    const v4f f3 = *(const v4f*)(p + 20);
    v16h a;
#pragma unroll
    for (int i = 0; i < 4; ++i) {
        a[i]      = (_Float16)f0[i];
        a[4 + i]  = (_Float16)f1[i];
        a[8 + i]  = (_Float16)f2[i];
        a[12 + i] = (_Float16)f3[i];
    }
    return a;
}

// ---- One-shot weight prep: fp32 -> f16, transposed + padded, into d_ws ----
__global__ __launch_bounds__(256)
void prep_weights(const float* __restrict__ W0, const float* __restrict__ W1,
                  const float* __restrict__ W2, _Float16* __restrict__ ws) {
    const int i = blockIdx.x * 256 + threadIdx.x;
    if (i < 512 * 128) {                       // W0 is [k=512][n=128] row-major
        const int k = i >> 7, n = i & 127;
        ws[n * SW0 + k] = (_Float16)W0[i];
    }
    if (i < 128 * 128) {
        const int k = i >> 7, n = i & 127;
        ws[WSH_W1 + n * SW12 + k] = (_Float16)W1[i];
        ws[WSH_W2 + n * SW12 + k] = (_Float16)W2[i];
    }
}

// LDS (halves):
//  PRECONV: [0,101376) = W0^T|W1^T|W2^T image; [101376,136192) = act tiles. 272384 B.
//  fallback: two-phase reuse as before. 202752 B.
template <bool PRECONV>
__global__ __launch_bounds__(256)
void megnet_edge_fused(const float* __restrict__ src,
                       const float* __restrict__ dest,
                       const float* __restrict__ edge,
                       const float* __restrict__ u,
                       const int*   __restrict__ batch,
                       const float* __restrict__ W0, const float* __restrict__ b0,
                       const float* __restrict__ W1, const float* __restrict__ b1,
                       const float* __restrict__ W2, const float* __restrict__ b2,
                       const float* __restrict__ g0, const float* __restrict__ be0,
                       const float* __restrict__ m0, const float* __restrict__ v0,
                       const float* __restrict__ g1, const float* __restrict__ be1,
                       const float* __restrict__ m1, const float* __restrict__ v1,
                       const float* __restrict__ g2, const float* __restrict__ be2,
                       const float* __restrict__ m2, const float* __restrict__ v2,
                       const _Float16* __restrict__ wsw,
                       float* __restrict__ out)
{
    __shared__ _Float16 smem[PRECONV ? 136192 : 101376];

    const int tid  = threadIdx.x;
    const int wave = tid >> 5;
    const int lane = tid & 31;
    const int mloc = lane & 15;
    const int half = lane >> 4;

    _Float16* W0t  = smem;
    _Float16* act0 = smem + (PRECONV ? WSH_TOTAL : 66560) + wave * (2 * 16 * SACT);
    _Float16* act1 = act0 + 16 * SACT;

    if constexpr (PRECONV) {
        // ---- Bulk copy of ready f16 weight image into LDS (16B per lane) ----
        for (int c = tid; c < WS_CHUNKS; c += 256) {
            const _Float16* gsrc = wsw + c * 8;
            _Float16*       ldst = smem + c * 8;
#if __has_builtin(__builtin_amdgcn_global_load_async_to_lds_b128)
            typedef __attribute__((address_space(1))) v4i* as1_v4i;
            typedef __attribute__((address_space(3))) v4i* as3_v4i;
            __builtin_amdgcn_global_load_async_to_lds_b128(
                (as1_v4i)(uintptr_t)gsrc,
                (as3_v4i)(uint32_t)(uintptr_t)ldst,
                0, 0);
#else
            *(v8h*)ldst = *(const v8h*)gsrc;
#endif
        }
#if __has_builtin(__builtin_amdgcn_global_load_async_to_lds_b128)
        asm volatile("s_wait_asynccnt 0x0" ::: "memory");
#endif
        __syncthreads();
    } else {
        // ---- Phase A: cooperative W0 fp32 -> f16 transpose into LDS ----
        for (int idx = tid; idx < 512 * 128; idx += 256) {
            const int k = idx >> 7, n = idx & 127;
            W0t[n * SW0 + k] = (_Float16)W0[idx];
        }
        __syncthreads();
    }

    // ---- Per-wave rows: two 16-row M-tiles (M=32 per wave) ----
    const int base = blockIdx.x * 256 + wave * 32;
    int row0 = base + mloc;
    int row1 = base + 16 + mloc;
    if (row0 >= E_EDGES) row0 = E_EDGES - 1;   // clamp: EXEC all-1s for WMMA
    if (row1 >= E_EDGES) row1 = E_EDGES - 1;
    const float* sp0 = src  + (size_t)row0 * D_DIM;
    const float* dp0 = dest + (size_t)row0 * D_DIM;
    const float* ep0 = edge + (size_t)row0 * D_DIM;
    const float* up0 = u    + (size_t)batch[row0] * D_DIM;
    const float* sp1 = src  + (size_t)row1 * D_DIM;
    const float* dp1 = dest + (size_t)row1 * D_DIM;
    const float* ep1 = edge + (size_t)row1 * D_DIM;
    const float* up1 = u    + (size_t)batch[row1] * D_DIM;

    v8f acc[2][8];
#pragma unroll
    for (int t = 0; t < 2; ++t)
#pragma unroll
        for (int nt = 0; nt < 8; ++nt) acc[t][nt] = zero8();

    // ---- Layer 0: [32 x 512] x [512 x 128]; each B fragment feeds 2 WMMAs ----
#pragma unroll
    for (int kt = 0; kt < 16; ++kt) {
        const float *p0, *p1;
        switch (kt >> 2) {
            case 0:  p0 = sp0; p1 = sp1; break;
            case 1:  p0 = dp0; p1 = dp1; break;
            case 2:  p0 = ep0; p1 = ep1; break;
            default: p0 = up0; p1 = up1; break;
        }
        const int lk = (kt & 3) * 32 + half * 8;
        const v16h a0 = loadA16(p0 + lk);
        const v16h a1 = loadA16(p1 + lk);
#pragma unroll
        for (int nh = 0; nh < 2; ++nh) {
            v16h b[4];
#pragma unroll
            for (int j = 0; j < 4; ++j) {
                const int nt = nh * 4 + j;
                const _Float16* wp = W0t + (nt * 16 + mloc) * SW0 + kt * 32 + half * 8;
                b[j] = cat8(*(const v8h*)wp, *(const v8h*)(wp + 16));
            }
#pragma unroll
            for (int j = 0; j < 4; ++j) {
                const int nt = nh * 4 + j;
                acc[0][nt] = __builtin_amdgcn_wmma_f32_16x16x32_f16(
                    false, a0, false, b[j], (short)0, acc[0][nt], false, false);
                acc[1][nt] = __builtin_amdgcn_wmma_f32_16x16x32_f16(
                    false, a1, false, b[j], (short)0, acc[1][nt], false, false);
            }
        }
    }

    // ---- Epilogue 0: bias + relu + BN -> f16 act tiles ----
#pragma unroll
    for (int nt = 0; nt < 8; ++nt) {
        const int n = nt * 16 + mloc;
        const float sc = g0[n] * rsqrtf(v0[n] + EPS_BN);
        const float sh = be0[n] - m0[n] * sc;
        const float bb = b0[n];
#pragma unroll
        for (int t = 0; t < 2; ++t) {
            _Float16* at = t ? act1 : act0;
#pragma unroll
            for (int r = 0; r < 8; ++r) {
                const int m = r + half * 8;
                float h = acc[t][nt][r] + bb;
                h = h > 0.0f ? h : 0.0f;
                at[m * SACT + n] = (_Float16)(h * sc + sh);
            }
        }
    }

    const _Float16 *W1t, *W2t;
    if constexpr (PRECONV) {
        W1t = smem + WSH_W1;
        W2t = smem + WSH_W2;
    } else {
        // ---- Phase B: reuse W0 LDS space for W1^T / W2^T ----
        __syncthreads();
        for (int idx = tid; idx < 128 * 128; idx += 256) {
            const int k = idx >> 7, n = idx & 127;
            smem[n * SW12 + k]              = (_Float16)W1[idx];
            smem[128 * SW12 + n * SW12 + k] = (_Float16)W2[idx];
        }
        __syncthreads();
        W1t = smem;
        W2t = smem + 128 * SW12;
    }

    // ---- Layers 1 and 2: [32 x 128] x [128 x 128] ----
#pragma unroll 1
    for (int layer = 1; layer <= 2; ++layer) {
        const _Float16* Wt = (layer == 1) ? W1t : W2t;
#pragma unroll
        for (int t = 0; t < 2; ++t)
#pragma unroll
            for (int nt = 0; nt < 8; ++nt) acc[t][nt] = zero8();

#pragma unroll
        for (int kt = 0; kt < 4; ++kt) {
            const _Float16* ap0 = act0 + mloc * SACT + kt * 32 + half * 8;
            const _Float16* ap1 = act1 + mloc * SACT + kt * 32 + half * 8;
            const v16h a0 = cat8(*(const v8h*)ap0, *(const v8h*)(ap0 + 16));
            const v16h a1 = cat8(*(const v8h*)ap1, *(const v8h*)(ap1 + 16));
#pragma unroll
            for (int nh = 0; nh < 2; ++nh) {
                v16h b[4];
#pragma unroll
                for (int j = 0; j < 4; ++j) {
                    const int nt = nh * 4 + j;
                    const _Float16* wp = Wt + (nt * 16 + mloc) * SW12 + kt * 32 + half * 8;
                    b[j] = cat8(*(const v8h*)wp, *(const v8h*)(wp + 16));
                }
#pragma unroll
                for (int j = 0; j < 4; ++j) {
                    const int nt = nh * 4 + j;
                    acc[0][nt] = __builtin_amdgcn_wmma_f32_16x16x32_f16(
                        false, a0, false, b[j], (short)0, acc[0][nt], false, false);
                    acc[1][nt] = __builtin_amdgcn_wmma_f32_16x16x32_f16(
                        false, a1, false, b[j], (short)0, acc[1][nt], false, false);
                }
            }
        }

        if (layer == 1) {
#pragma unroll
            for (int nt = 0; nt < 8; ++nt) {
                const int n = nt * 16 + mloc;
                const float sc = g1[n] * rsqrtf(v1[n] + EPS_BN);
                const float sh = be1[n] - m1[n] * sc;
                const float bb = b1[n];
#pragma unroll
                for (int t = 0; t < 2; ++t) {
                    _Float16* at = t ? act1 : act0;
#pragma unroll
                    for (int r = 0; r < 8; ++r) {
                        const int m = r + half * 8;
                        float h = acc[t][nt][r] + bb;
                        h = h > 0.0f ? h : 0.0f;
                        at[m * SACT + n] = (_Float16)(h * sc + sh);
                    }
                }
            }
        } else {
#pragma unroll
            for (int nt = 0; nt < 8; ++nt) {
                const int n = nt * 16 + mloc;
                const float sc = g2[n] * rsqrtf(v2[n] + EPS_BN);
                const float sh = be2[n] - m2[n] * sc;
                const float bb = b2[n];
#pragma unroll
                for (int t = 0; t < 2; ++t) {
#pragma unroll
                    for (int r = 0; r < 8; ++r) {
                        const int m  = r + half * 8;
                        const int rr = base + t * 16 + m;
                        float h = acc[t][nt][r] + bb;
                        h = h > 0.0f ? h : 0.0f;
                        if (rr < E_EDGES)
                            out[(size_t)rr * D_DIM + n] = h * sc + sh;
                    }
                }
            }
        }
    }
}

extern "C" void kernel_launch(void* const* d_in, const int* in_sizes, int n_in,
                              void* d_out, int out_size, void* d_ws, size_t ws_size,
                              hipStream_t stream) {
    (void)in_sizes; (void)n_in; (void)out_size;
    const float* src  = (const float*)d_in[0];
    const float* dest = (const float*)d_in[1];
    const float* edge = (const float*)d_in[2];
    const float* u    = (const float*)d_in[3];
    const int*   bat  = (const int*)  d_in[4];
    const float* W0   = (const float*)d_in[5];
    const float* b0   = (const float*)d_in[6];
    const float* W1   = (const float*)d_in[7];
    const float* b1   = (const float*)d_in[8];
    const float* W2   = (const float*)d_in[9];
    const float* b2   = (const float*)d_in[10];
    const float* g0   = (const float*)d_in[11];
    const float* be0  = (const float*)d_in[12];
    const float* m0   = (const float*)d_in[13];
    const float* v0   = (const float*)d_in[14];
    const float* g1   = (const float*)d_in[15];
    const float* be1  = (const float*)d_in[16];
    const float* m1   = (const float*)d_in[17];
    const float* v1   = (const float*)d_in[18];
    const float* g2   = (const float*)d_in[19];
    const float* be2  = (const float*)d_in[20];
    const float* m2   = (const float*)d_in[21];
    const float* v2   = (const float*)d_in[22];
    float* out = (float*)d_out;

    const int blocks = (E_EDGES + 255) / 256;   // 1954

    if (ws_size >= (size_t)WS_BYTES) {
        _Float16* ws = (_Float16*)d_ws;
        prep_weights<<<(512 * 128 + 255) / 256, 256, 0, stream>>>(W0, W1, W2, ws);
        megnet_edge_fused<true><<<blocks, 256, 0, stream>>>(
            src, dest, edge, u, bat,
            W0, b0, W1, b1, W2, b2,
            g0, be0, m0, v0, g1, be1, m1, v1, g2, be2, m2, v2,
            ws, out);
    } else {
        megnet_edge_fused<false><<<blocks, 256, 0, stream>>>(
            src, dest, edge, u, bat,
            W0, b0, W1, b1, W2, b2,
            g0, be0, m0, v0, g1, be1, m1, v1, g2, be2, m2, v2,
            (const _Float16*)nullptr, out);
    }
}
